// RelativeAgentPositionEmbedding_32890859552915
// MI455X (gfx1250) — matrix operand order repaired and checked by reading, not compiled, not run on hardware
//
#include <hip/hip_runtime.h>

#define N_NODES 50000
#define N_EDGES 500000
#define NDIM 6
#define NTILES (N_EDGES / 16)
#define WAVES 4

#define PI_F     3.14159265358979323846f
#define TWO_PI_F 6.28318530717958647692f

typedef __attribute__((ext_vector_type(16))) _Float16 v16h;
typedef __attribute__((ext_vector_type(8)))  float    v8f;
typedef __attribute__((ext_vector_type(4)))  float    f4;

// wave-local LDS ordering fence (all producers/consumers are in this wave)
__device__ __forceinline__ void lds_fence() {
    asm volatile("s_wait_dscnt 0" ::: "memory");
}

// hardware sin/cos compute sin/cos(2*pi*x); use directly to avoid the
// (x*2pi)*(1/2pi) multiply pair that __sinf/__cosf would emit.
__device__ __forceinline__ float sin_2pi(float turns) {
#if __has_builtin(__builtin_amdgcn_sinf)
    return __builtin_amdgcn_sinf(turns);
#else
    return __sinf(turns * TWO_PI_F);
#endif
}
__device__ __forceinline__ float cos_2pi(float turns) {
#if __has_builtin(__builtin_amdgcn_cosf)
    return __builtin_amdgcn_cosf(turns);
#else
    return __cosf(turns * TWO_PI_F);
#endif
}

__device__ __forceinline__ float wrap_angle_f(float a) {
    float r = fmodf(a + PI_F, TWO_PI_F);
    if (r < 0.f) r += TWO_PI_F;
    return r - PI_F;
}

// ---------------------------------------------------------------------------
// Prep: repack weights into WMMA-B-fragment-friendly transposed f16 layouts.
//   w1t   [d][n][p] p=0..63, k=p w/ bits 3,4 swapped  (f16)   <- W1[d][k][n]           (98,304 B)
//   w1last[d][n]             (f32)   <- W1[d][64][n]          (  3,072 B)
//   w2t   [d][n][p] p=0..127, k=p w/ bits 3,4 swapped (f16)   <- W2[d][k][n]           (196,608 B)
//   b2sum [n]                (f32)   <- sum_d b2[d][n]        (    512 B)
// ---------------------------------------------------------------------------
__global__ __launch_bounds__(256) void prep_kernel(
    const float* __restrict__ W1, const float* __restrict__ W2,
    const float* __restrict__ b2,
    _Float16* __restrict__ w1t, float* __restrict__ w1last,
    _Float16* __restrict__ w2t, float* __restrict__ b2sum) {
    int idx = blockIdx.x * blockDim.x + threadIdx.x;
    if (idx < 49152) {                       // w1t
        int d = idx >> 13;
        int n = (idx >> 6) & 127;
        int k = (idx & 39) | ((idx >> 1) & 8) | ((idx << 1) & 16);
        w1t[idx] = (_Float16)W1[(d * 65 + k) * 128 + n];
    } else if (idx < 49920) {                // w1last (K=64 row, kept f32)
        int i = idx - 49152;
        int d = i >> 7, n = i & 127;
        w1last[i] = W1[(d * 65 + 64) * 128 + n];
    } else if (idx < 148224) {               // w2t
        int i = idx - 49920;
        int d = i >> 14;
        int n = (i >> 7) & 127;
        int k = (i & 103) | ((i >> 1) & 8) | ((i << 1) & 16);
        w2t[i] = (_Float16)W2[(d * 128 + k) * 128 + n];
    } else if (idx < 148352) {               // b2sum
        int n = idx - 148224;
        float s = 0.f;
        for (int d = 0; d < NDIM; ++d) s += b2[d * 128 + n];
        b2sum[n] = s;
    }
}

// ---------------------------------------------------------------------------
// Fused kernel: one wave = one 16-edge tile.
//   feat -> fourier A-frags -> WMMA GEMM1 (K=64, C preloaded with b1+rank1)
//   -> LDS h stage -> LayerNorm (shfl_xor cross-half) -> ReLU -> f16 A-frags
//   -> WMMA GEMM2 (K=128) accumulated over the 6 input dims -> out (+b2sum)
// ---------------------------------------------------------------------------
__global__ __launch_bounds__(WAVES * 32) void fused_kernel(
    const float* __restrict__ source, const float* __restrict__ target,
    const long long* __restrict__ edge, const float* __restrict__ freqs,
    const float* __restrict__ b1, const float* __restrict__ ln_w,
    const float* __restrict__ ln_b,
    const _Float16* __restrict__ w1t, const float* __restrict__ w1last,
    const _Float16* __restrict__ w2t, const float* __restrict__ b2sum,
    float* __restrict__ out) {
    __shared__ float ldsH[WAVES][16 * 128];   // per-wave h stage (f32)
    __shared__ float ldsFeat[WAVES][16 * 6];  // per-wave features
    __shared__ float ldsStat[WAVES][32];      // mu[16], rsig[16]

    const int lane = threadIdx.x & 31;
    const int w    = threadIdx.x >> 5;
    const int tile = blockIdx.x * WAVES + w;
    if (tile >= NTILES) return;               // wave-uniform
    const int e0  = tile * 16;
    const int nl  = lane & 15;                // column / row selector
    const int hi  = lane >> 4;                // half-wave id
    const int off = hi << 3;                  // A-frag K / C-frag M offset

    // ---- features: lanes 0..15 each handle one edge ----
    if (lane < 16) {
        const int e = e0 + lane;
        const long long si = edge[e];
        const long long ti = edge[N_EDGES + e];
        const float* S = source + si * 5;
        const float* T = target + ti * 5;
        float spx = S[0], spy = S[1], sh = S[2], ss = S[3];
        float tpx = T[0], tpy = T[1], th = T[2], ts = T[3];
        float dx = spx - tpx, dy = spy - tpy;
        float dist = sqrtf(dx * dx + dy * dy);
        float is_close = (dist < 3.0f) ? 1.0f : 0.0f;
        float ang = atan2f(dy, dx);
        float dh = wrap_angle_f(sh - th);
        float ca = cosf(ang), sa = sinf(ang);
        float t_close = ts * cosf(th) * ca + ts * sinf(th) * sa;
        float s_close = ss * cosf(sh) * ca + ss * sinf(sh) * sa;
        float* fr = &ldsFeat[w][lane * 6];
        fr[0] = dx; fr[1] = dy; fr[2] = dh;
        fr[3] = is_close; fr[4] = t_close; fr[5] = s_close;
    }
    lds_fence();

    // ---- output accumulators: 8 N-tiles of 16x16 f32 C frags, init b2sum ----
    v8f acc[8];
    #pragma unroll
    for (int nt = 0; nt < 8; ++nt) {
        float b2v = b2sum[nt * 16 + nl];
        #pragma unroll
        for (int v = 0; v < 8; ++v) acc[nt][v] = b2v;
    }

    for (int d = 0; d < NDIM; ++d) {
        __builtin_prefetch(w2t + d * 16384, 0, 1);

        // ---- GEMM1 A-frags: cos/sin(2pi * feat * freq) in the ISA lane map --
        // lane holds row M=nl; K = s*32 + off + (j<8 ? j : j+8)
        float frq[16];
        {
            const float* fp = freqs + d * 32 + off;
            *(f4*)(frq + 0)  = *(const f4*)(fp + 0);
            *(f4*)(frq + 4)  = *(const f4*)(fp + 4);
            *(f4*)(frq + 8)  = *(const f4*)(fp + 16);
            *(f4*)(frq + 12) = *(const f4*)(fp + 20);
        }
        const float fm = ldsFeat[w][nl * 6 + d];
        v16h a0, a1;
        #pragma unroll
        for (int j = 0; j < 16; ++j) {
            float t = fm * frq[j];               // in turns (units of 2*pi)
            a0[j] = (_Float16)cos_2pi(t);        // K-step 0: cos block (k<32)
            a1[j] = (_Float16)sin_2pi(t);        // K-step 1: sin block (k>=32)
        }
        float fm8[8];
        #pragma unroll
        for (int v = 0; v < 8; ++v) fm8[v] = ldsFeat[w][(v + off) * 6 + d];

        // ---- GEMM1: h = x @ W1[d] + (b1 + feat*W1[d,64,:]) via C preload ----
        #pragma unroll
        for (int nt = 0; nt < 8; ++nt) {
            const int n = nt * 16 + nl;
            const v16h* bp = (const v16h*)(w1t + (size_t)(d * 128 + n) * 64 + hi * 16);
            const float bb = b1[d * 128 + n];
            const float wl = w1last[d * 128 + n];
            v8f c;
            #pragma unroll
            for (int v = 0; v < 8; ++v) c[v] = fmaf(fm8[v], wl, bb);
            c = __builtin_amdgcn_wmma_f32_16x16x32_f16(false, a0, false, bp[0],
                                                       (short)0, c, false, false);
            c = __builtin_amdgcn_wmma_f32_16x16x32_f16(false, a1, false, bp[2],
                                                       (short)0, c, false, false);
            #pragma unroll
            for (int v = 0; v < 8; ++v)
                ldsH[w][(v + off) * 128 + n] = c[v];
        }
        lds_fence();

        // ---- LayerNorm stats: lane = (row nl, half hi), 64 cols each ----
        {
            const float* hp = &ldsH[w][nl * 128 + hi * 64];
            float sm = 0.f, sq = 0.f;
            #pragma unroll
            for (int i = 0; i < 64; i += 4) {
                f4 hv = *(const f4*)(hp + i);
                #pragma unroll
                for (int t = 0; t < 4; ++t) { sm += hv[t]; sq += hv[t] * hv[t]; }
            }
            sm += __shfl_xor(sm, 16, 32);
            sq += __shfl_xor(sq, 16, 32);
            float mu  = sm * (1.0f / 128.0f);
            float var = sq * (1.0f / 128.0f) - mu * mu;
            float rs  = rsqrtf(var + 1e-5f);
            ldsStat[w][nl]      = mu;
            ldsStat[w][16 + nl] = rs;
        }
        lds_fence();

        // ---- GEMM2 A-frags: normalize + scale/shift + ReLU -> f16 ----
        const float mu = ldsStat[w][nl];
        const float rs = ldsStat[w][16 + nl];
        v16h aF[4];
        #pragma unroll
        for (int ks = 0; ks < 4; ++ks) {
            const int kb = ks * 32 + off;
            const float* hp  = &ldsH[w][nl * 128 + kb];
            const float* lwp = ln_w + d * 128 + kb;
            const float* lbp = ln_b + d * 128 + kb;
            f4 hq[4] = { *(const f4*)(hp + 0),  *(const f4*)(hp + 4),
                         *(const f4*)(hp + 16), *(const f4*)(hp + 20) };
            f4 wq[4] = { *(const f4*)(lwp + 0),  *(const f4*)(lwp + 4),
                         *(const f4*)(lwp + 16), *(const f4*)(lwp + 20) };
            f4 bq[4] = { *(const f4*)(lbp + 0),  *(const f4*)(lbp + 4),
                         *(const f4*)(lbp + 16), *(const f4*)(lbp + 20) };
            #pragma unroll
            for (int j = 0; j < 16; ++j) {
                float val = (hq[j >> 2][j & 3] - mu) * rs * wq[j >> 2][j & 3]
                            + bq[j >> 2][j & 3];
                aF[ks][j] = (_Float16)fmaxf(val, 0.0f);
            }
        }

        // ---- GEMM2: out += relu(LN(h)) @ W2[d] ----
        #pragma unroll
        for (int nt = 0; nt < 8; ++nt) {
            const int n = nt * 16 + nl;
            const v16h* bp = (const v16h*)(w2t + (size_t)(d * 128 + n) * 128 + hi * 16);
            #pragma unroll
            for (int ks = 0; ks < 4; ++ks) {
                acc[nt] = __builtin_amdgcn_wmma_f32_16x16x32_f16(
                    false, aF[ks], false, bp[ks * 2], (short)0, acc[nt], false, false);
            }
        }
    }

    // ---- store: C layout (M = v + 8*hi, N = nt*16 + nl) ----
    #pragma unroll
    for (int nt = 0; nt < 8; ++nt) {
        #pragma unroll
        for (int v = 0; v < 8; ++v) {
            out[(size_t)(e0 + v + off) * 128 + nt * 16 + nl] = acc[nt][v];
        }
    }
}

extern "C" void kernel_launch(void* const* d_in, const int* in_sizes, int n_in,
                              void* d_out, int out_size, void* d_ws, size_t ws_size,
                              hipStream_t stream) {
    (void)in_sizes; (void)n_in; (void)out_size; (void)ws_size;
    const float*     source = (const float*)d_in[0];
    const float*     target = (const float*)d_in[1];
    const long long* edge   = (const long long*)d_in[2];
    const float*     freqs  = (const float*)d_in[3];
    const float*     W1     = (const float*)d_in[4];
    const float*     b1     = (const float*)d_in[5];
    const float*     ln_w   = (const float*)d_in[6];
    const float*     ln_b   = (const float*)d_in[7];
    const float*     W2     = (const float*)d_in[8];
    const float*     b2     = (const float*)d_in[9];
    float*           out    = (float*)d_out;

    char* ws = (char*)d_ws;
    _Float16* w1t    = (_Float16*)(ws + 0);       //  98,304 B
    float*    w1last = (float*)   (ws + 98304);   //   3,072 B
    _Float16* w2t    = (_Float16*)(ws + 101376);  // 196,608 B
    float*    b2sumv = (float*)   (ws + 297984);  //     512 B

    prep_kernel<<<(148352 + 255) / 256, 256, 0, stream>>>(
        W1, W2, b2, w1t, w1last, w2t, b2sumv);
    fused_kernel<<<(NTILES + WAVES - 1) / WAVES, WAVES * 32, 0, stream>>>(
        source, target, edge, freqs, b1, ln_w, ln_b,
        w1t, w1last, w2t, b2sumv, out);
}
